// ExceptionalEGNN_3977139716371
// MI455X (gfx1250) — matrix operand order repaired and checked by reading, not compiled, not run on hardware
//
#include <hip/hip_runtime.h>

#define D_    248
#define HID_  256
#define IN_   128
#define OUT_  16
#define NL_   4
#define N_    20000
#define E_    100000
#define NNZ_  512
#define NB_   1024
#define NG_   128

typedef __bf16 bf16;
typedef __attribute__((ext_vector_type(16))) __bf16    v16bf;
typedef __attribute__((ext_vector_type(8)))  float     v8f;
typedef __attribute__((ext_vector_type(8)))  unsigned  v8u;

__device__ __forceinline__ unsigned short f2bf_raw(float f) {
  unsigned u = __builtin_bit_cast(unsigned, f);
  unsigned r = u + 0x7FFFu + ((u >> 16) & 1u);   // round-to-nearest-even
  return (unsigned short)(r >> 16);
}

// Pack two fp32 -> packed bf16 pair (lo = a, hi = b) with round-half-up bias:
// one v_perm_b32 selecting the high halves after +0x8000.
__device__ __forceinline__ unsigned pack_bf16(float a, float b) {
  unsigned ua = __builtin_bit_cast(unsigned, a) + 0x8000u;
  unsigned ub = __builtin_bit_cast(unsigned, b) + 0x8000u;
  return __builtin_amdgcn_perm(ub, ua, 0x07060302u);
}

// Fast silu: v_exp + v_add + v_rcp + v_mul (avoids IEEE div expansion).
__device__ __forceinline__ float silu_f(float x) {
  return x * __builtin_amdgcn_rcpf(1.0f + __expf(-x));
}

// ---------------------------------------------------------------------------
// Weight conversion: fp32 W[K,Nout] -> bf16 fragment-major buffer.
// Wb[(kt*Npc + np)*256 + lane*8 + e] packs B-frag elems (2e, 2e+1):
//   k = kt*32 + (lane>>4)*16 + 2e,  n = np*16 + (lane&15).  Zero-padded.
// ---------------------------------------------------------------------------
__global__ void convert_w_kernel(const float* __restrict__ W, unsigned* __restrict__ Wb,
                                 int K, int Nout, int Npc, int lnp, int total) {
  int i = blockIdx.x * blockDim.x + threadIdx.x;
  if (i >= total) return;
  int e    = i & 7;
  int lane = (i >> 3) & 31;
  int rest = i >> 8;
  int np   = rest & (Npc - 1);
  int kt   = rest >> lnp;
  int k = kt * 32 + ((lane >> 4) << 4) + 2 * e;
  int n = np * 16 + (lane & 15);
  float f0 = (k < K && n < Nout) ? W[k * Nout + n] : 0.0f;
  float f1 = (k + 1 < K && n < Nout) ? W[(k + 1) * Nout + n] : 0.0f;
  Wb[i] = (unsigned)f2bf_raw(f0) | ((unsigned)f2bf_raw(f1) << 16);
}

// ---------------------------------------------------------------------------
// WMMA GEMM: C[M,Nout] = act( concat(A0[gather, :K0], A1[:, K0:]) @ W + bias )
// Block = 256 threads (8 waves), C-tile 128x64; each wave owns 32x32 (4 WMMA).
// Branch-free staging: row/col clamps instead of guards (Wb is zero-padded,
// so out-of-range A values are multiplied by zero). K must be a multiple of 4.
// AMODE: 0 = plain A0, 1 = concat(A0,A1), 2 = concat(A0[idx0[row]], A1[row])
// EPI:   0 = store, 1 = accumulate, 2 = atomicAdd into C[scatter[m]]
// ---------------------------------------------------------------------------
template<int AMODE, int EPI, int ACT>
__global__ __launch_bounds__(256)
void wmma_gemm_kernel(const float* __restrict__ A0, const float* __restrict__ A1,
                      const int* __restrict__ idx0, int lda0, int lda1, int K0,
                      const unsigned* __restrict__ Wb, int Npc,
                      const float* __restrict__ bias,
                      float* __restrict__ C, int ldc, const int* __restrict__ scatter,
                      int M, int K, int Nout) {
  // A tile in WMMA fragment order: [m-subtile][lane][elem-pair]
  __shared__ unsigned AsF[8][32][8];

  const int tid  = threadIdx.x;
  const int lane = tid & 31;
  const int wv   = tid >> 5;
  const int mblk = blockIdx.x * 128;
  const int nblk = blockIdx.y * 64;
  const int wm   = (wv >> 1) << 5;   // 0,32,64,96
  const int wn   = (wv & 1) << 5;    // 0,32

  // Staging: thread loads fp32 quad at (row = srow+32t, col = qc..qc+3)
  const int srow = tid >> 3;          // 0..31
  const int qc   = (tid & 7) << 2;    // 0,4,...,28
  const int dlane = (((qc >> 4) & 1) << 4) + (srow & 15);
  const int delem = (qc & 15) >> 1;   // even: 0,2,4,6

  // kt-invariant row offsets (clamped) and LDS store pointers
  int g0[4], g1[4];
  unsigned long long* dstp[4];
#pragma unroll
  for (int t = 0; t < 4; ++t) {
    int row = mblk + srow + t * 32;
    int rc = row < M - 1 ? row : M - 1;
    int r0 = rc;
    if (AMODE == 2) r0 = idx0[rc];
    g0[t] = r0 * lda0;
    g1[t] = rc * lda1;
    dstp[t] = (unsigned long long*)&AsF[(srow >> 4) + 2 * t][dlane][delem];
  }

  v8f acc[2][2] = {};

  for (int kt = 0; kt < K; kt += 32) {
    // ---- stage A tile: 1x global b128 + 2x v_perm + 1x ds b64 per quad ----
#pragma unroll
    for (int t = 0; t < 4; ++t) {
      int gk = kt + qc;
      int gkc = gk < K - 4 ? gk : K - 4;   // zero-weighted beyond K
      const float* p;
      if (AMODE == 0) p = A0 + g0[t] + gkc;
      else            p = (gkc < K0) ? (A0 + g0[t] + gkc)
                                     : (A1 + g1[t] + (gkc - K0));
      float4 v = *(const float4*)p;
      unsigned lo = pack_bf16(v.x, v.y);
      unsigned hi = pack_bf16(v.z, v.w);
      *dstp[t] = (unsigned long long)lo | ((unsigned long long)hi << 32);
    }
    __syncthreads();

    // ---- B fragments: wide loads from converted weights (L2 resident) ----
    const int ktb = (kt >> 5) * Npc;
    const int np0 = (nblk + wn) >> 4;
    v8u br0 = *(const v8u*)(Wb + (((ktb + np0) << 8) | (lane << 3)));
    v8u br1 = *(const v8u*)(Wb + (((ktb + np0 + 1) << 8) | (lane << 3)));
    v16bf b0 = __builtin_bit_cast(v16bf, br0);
    v16bf b1 = __builtin_bit_cast(v16bf, br1);

    // ---- A fragments: 2x ds_load_b128 each ----
    const int sm = wm >> 4;
    v8u ar0 = *(const v8u*)&AsF[sm][lane][0];
    v8u ar1 = *(const v8u*)&AsF[sm + 1][lane][0];
    v16bf a0 = __builtin_bit_cast(v16bf, ar0);
    v16bf a1 = __builtin_bit_cast(v16bf, ar1);

    acc[0][0] = __builtin_amdgcn_wmma_f32_16x16x32_bf16(false, a0, false, b0, (short)0, acc[0][0], false, false);
    acc[0][1] = __builtin_amdgcn_wmma_f32_16x16x32_bf16(false, a0, false, b1, (short)0, acc[0][1], false, false);
    acc[1][0] = __builtin_amdgcn_wmma_f32_16x16x32_bf16(false, a1, false, b0, (short)0, acc[1][0], false, false);
    acc[1][1] = __builtin_amdgcn_wmma_f32_16x16x32_bf16(false, a1, false, b1, (short)0, acc[1][1], false, false);
    __syncthreads();
  }

  // ---- epilogue: C layout VGPR v -> M = v + 8*(lane>=16), N = lane&15 ----
#pragma unroll
  for (int j = 0; j < 2; ++j) {
    int cn = nblk + wn + j * 16 + (lane & 15);
    if (cn < Nout) {
      float bv = bias[cn];
#pragma unroll
      for (int i = 0; i < 2; ++i) {
#pragma unroll
        for (int v = 0; v < 8; ++v) {
          int m = mblk + wm + i * 16 + v + ((lane >> 4) << 3);
          if (m < M) {
            float val = acc[i][j][v] + bv;
            if (ACT) val = silu_f(val);
            if (EPI == 0)      C[m * ldc + cn] = val;
            else if (EPI == 1) C[m * ldc + cn] += val;
            else               atomicAdd(&C[scatter[m] * ldc + cn], val);
          }
        }
      }
    }
  }
}

// ---------------------------------------------------------------------------
// Lie bracket via sparse structure constants; one block per edge, LDS atomics.
// ---------------------------------------------------------------------------
__global__ __launch_bounds__(256)
void bracket_kernel(const float* __restrict__ h, const int* __restrict__ src,
                    const int* __restrict__ dst,
                    const int* __restrict__ sci, const int* __restrict__ scj,
                    const int* __restrict__ sck, const float* __restrict__ scc,
                    float* __restrict__ br) {
  __shared__ float hs[D_], hd[D_], lbr[D_];
  int e = blockIdx.x;
  int tid = threadIdx.x;
  int s = src[e], d = dst[e];
  for (int c = tid; c < D_; c += 256) {
    hs[c] = h[s * D_ + c];
    hd[c] = h[d * D_ + c];
    lbr[c] = 0.0f;
  }
  __syncthreads();
  for (int z = tid; z < NNZ_; z += 256) {
    float v = scc[z] * hs[sci[z]] * hd[scj[z]];
    atomicAdd(&lbr[sck[z]], v);   // ds_add_f32
  }
  __syncthreads();
  for (int c = tid; c < D_; c += 256) br[e * D_ + c] = lbr[c];
}

// Killing-form invariant + mean-pool scatter; one block per node.
__global__ __launch_bounds__(256)
void pool_kernel(const float* __restrict__ h, const int* __restrict__ batch,
                 const int* __restrict__ kbr, const int* __restrict__ kbc,
                 const float* __restrict__ kbv,
                 float* __restrict__ pooled, float* __restrict__ counts) {
  __shared__ float hrow[D_];
  __shared__ float red[256];
  int n = blockIdx.x, tid = threadIdx.x;
  for (int c = tid; c < D_; c += 256) hrow[c] = h[n * D_ + c];
  __syncthreads();
  float acc = 0.0f;
  for (int z = tid; z < NB_; z += 256) acc += kbv[z] * hrow[kbr[z]] * hrow[kbc[z]];
  red[tid] = acc;
  __syncthreads();
  for (int s = 128; s > 0; s >>= 1) {
    if (tid < s) red[tid] += red[tid + s];
    __syncthreads();
  }
  int g = batch[n];
  for (int c = tid; c < D_; c += 256) atomicAdd(&pooled[g * 256 + c], hrow[c]);
  if (tid == 0) {
    atomicAdd(&pooled[g * 256 + D_], red[0]);
    atomicAdd(&counts[g], 1.0f);
  }
}

__global__ void normalize_kernel(float* __restrict__ pooled,
                                 const float* __restrict__ counts) {
  int i = blockIdx.x * blockDim.x + threadIdx.x;
  if (i < NG_ * 256) pooled[i] /= fmaxf(counts[i >> 8], 1.0f);
}

__global__ void fill_kernel(float* __restrict__ p, int n) {
  int i = blockIdx.x * blockDim.x + threadIdx.x;
  int stride = gridDim.x * blockDim.x;
  for (; i < n; i += stride) p[i] = 0.0f;
}

extern "C" void kernel_launch(void* const* d_in, const int* in_sizes, int n_in,
                              void* d_out, int out_size, void* d_ws, size_t ws_size,
                              hipStream_t stream) {
  (void)in_sizes; (void)n_in; (void)out_size; (void)ws_size;
  const float* x     = (const float*)d_in[0];
  const int*   eidx  = (const int*)d_in[1];
  const int*   batch = (const int*)d_in[2];
  const int*   sc_i  = (const int*)d_in[3];
  const int*   sc_j  = (const int*)d_in[4];
  const int*   sc_k  = (const int*)d_in[5];
  const float* sc_c  = (const float*)d_in[6];
  const int*   kb_r  = (const int*)d_in[7];
  const int*   kb_c  = (const int*)d_in[8];
  const float* kb_v  = (const float*)d_in[9];
  const float* Wi1   = (const float*)d_in[10];
  const float* bi1   = (const float*)d_in[11];
  const float* Wi2   = (const float*)d_in[12];
  const float* bi2   = (const float*)d_in[13];
  const float* msgW1 = (const float*)d_in[14];
  const float* msgb1 = (const float*)d_in[15];
  const float* msgW2 = (const float*)d_in[16];
  const float* msgb2 = (const float*)d_in[17];
  const float* updW1 = (const float*)d_in[18];
  const float* updb1 = (const float*)d_in[19];
  const float* updW2 = (const float*)d_in[20];
  const float* updb2 = (const float*)d_in[21];
  const float* Wo1   = (const float*)d_in[22];
  const float* bo1   = (const float*)d_in[23];
  const float* Wo2   = (const float*)d_in[24];
  const float* bo2   = (const float*)d_in[25];
  float* out = (float*)d_out;

  float* ws = (float*)d_ws;
  size_t off = 0;
  float* t0     = ws + off; off += (size_t)N_ * HID_;
  float* h      = ws + off; off += (size_t)N_ * D_;
  float* br     = ws + off; off += (size_t)E_ * D_;
  float* eh     = ws + off; off += (size_t)E_ * HID_;
  float* agg    = ws + off; off += (size_t)N_ * D_;
  float* pooled = ws + off; off += (size_t)NG_ * 256;
  float* counts = ws + off; off += (size_t)NG_;
  float* pg     = ws + off; off += (size_t)NG_ * HID_;
  off = (off + 15) & ~(size_t)15;                 // 64B-align weight region
  unsigned* wp = (unsigned*)(ws + off);

  dim3 blk(256);
  auto npcOf = [](int Nout) { return 4 * ((Nout + 63) / 64); };
  auto convW = [&](const float* W, int K, int Nout) -> unsigned* {
    int Npc = npcOf(Nout);
    int lnp = (Npc == 16) ? 4 : 2;
    int Kp = (K + 31) / 32;
    int total = Kp * Npc * 256;
    unsigned* dst = wp;
    convert_w_kernel<<<(total + 255) / 256, blk, 0, stream>>>(W, dst, K, Nout, Npc, lnp, total);
    wp += total;
    return dst;
  };

  // Convert all weights to bf16 fragment-major once (L2-resident, ~3.5 MB).
  unsigned* cWi1 = convW(Wi1, IN_, HID_);
  unsigned* cWi2 = convW(Wi2, HID_, D_);
  unsigned *cm1[NL_], *cm2[NL_], *cu1[NL_], *cu2[NL_];
  for (int l = 0; l < NL_; ++l) {
    cm1[l] = convW(msgW1 + (size_t)l * 2 * D_ * HID_, 2 * D_, HID_);
    cm2[l] = convW(msgW2 + (size_t)l * HID_ * D_, HID_, D_);
    cu1[l] = convW(updW1 + (size_t)l * 2 * D_ * HID_, 2 * D_, HID_);
    cu2[l] = convW(updW2 + (size_t)l * HID_ * D_, HID_, D_);
  }
  unsigned* cWo1 = convW(Wo1, D_ + 1, HID_);
  unsigned* cWo2 = convW(Wo2, HID_, OUT_);

  const int* src = eidx;
  const int* dst = eidx + E_;
  auto g = [](int M, int Nout) { return dim3((M + 127) / 128, (Nout + 63) / 64); };

  // Input MLP
  wmma_gemm_kernel<0, 0, 1><<<g(N_, HID_), blk, 0, stream>>>(
      x, nullptr, nullptr, IN_, 0, IN_, cWi1, 16, bi1, t0, HID_, nullptr, N_, IN_, HID_);
  wmma_gemm_kernel<0, 0, 0><<<g(N_, D_), blk, 0, stream>>>(
      t0, nullptr, nullptr, HID_, 0, HID_, cWi2, 16, bi2, h, D_, nullptr, N_, HID_, D_);

  for (int l = 0; l < NL_; ++l) {
    const float* mb1 = msgb1 + (size_t)l * HID_;
    const float* mb2 = msgb2 + (size_t)l * D_;
    const float* ub1 = updb1 + (size_t)l * HID_;
    const float* ub2 = updb2 + (size_t)l * D_;

    bracket_kernel<<<E_, blk, 0, stream>>>(h, src, dst, sc_i, sc_j, sc_k, sc_c, br);

    // msg hidden: silu(concat(h[src], br) @ mW1 + mb1); gather fused into A-tile
    wmma_gemm_kernel<2, 0, 1><<<g(E_, HID_), blk, 0, stream>>>(
        h, br, src, D_, D_, D_, cm1[l], 16, mb1, eh, HID_, nullptr, E_, 2 * D_, HID_);

    fill_kernel<<<2048, blk, 0, stream>>>(agg, N_ * D_);

    // m = eh @ mW2 + mb2 with fused scatter-add into agg[dst[e]]
    wmma_gemm_kernel<0, 2, 0><<<g(E_, D_), blk, 0, stream>>>(
        eh, nullptr, nullptr, HID_, 0, HID_, cm2[l], 16, mb2, agg, D_, dst, E_, HID_, D_);

    // update hidden: silu(concat(h, agg) @ uW1 + ub1)
    wmma_gemm_kernel<1, 0, 1><<<g(N_, HID_), blk, 0, stream>>>(
        h, agg, nullptr, D_, D_, D_, cu1[l], 16, ub1, t0, HID_, nullptr, N_, 2 * D_, HID_);

    // h += t0 @ uW2 + ub2
    wmma_gemm_kernel<0, 1, 0><<<g(N_, D_), blk, 0, stream>>>(
        t0, nullptr, nullptr, HID_, 0, HID_, cu2[l], 16, ub2, h, D_, nullptr, N_, HID_, D_);
  }

  // Killing-form scalar + mean pool per graph
  fill_kernel<<<64, blk, 0, stream>>>(pooled, NG_ * 256 + NG_);
  pool_kernel<<<N_, blk, 0, stream>>>(h, batch, kb_r, kb_c, kb_v, pooled, counts);
  normalize_kernel<<<(NG_ * 256 + 255) / 256, blk, 0, stream>>>(pooled, counts);

  // Output MLP: promote K to 256 (pooled cols 249..255 are zero, Wb zero-padded)
  wmma_gemm_kernel<0, 0, 1><<<g(NG_, HID_), blk, 0, stream>>>(
      pooled, nullptr, nullptr, 256, 0, 256, cWo1, 16, bo1, pg, HID_, nullptr,
      NG_, 256, HID_);
  wmma_gemm_kernel<0, 0, 0><<<g(NG_, OUT_), blk, 0, stream>>>(
      pg, nullptr, nullptr, HID_, 0, HID_, cWo2, 4, bo2, out, OUT_, nullptr,
      NG_, HID_, OUT_);
}
